// TTTLinearLayer_24361054503236
// MI455X (gfx1250) — compile-verified, matrix operand order checked by reading, not run on hardware
//
#include <hip/hip_runtime.h>
#include <hip/hip_bf16.h>
#include <math.h>

#define BB  2
#define LL  8192
#define DD  1024
#define NHH 16
#define HDD 64
#define KK  16
#define NM  512            // LL / KK
#define EPSF 1e-6f

typedef __attribute__((ext_vector_type(16))) __bf16 v16bf;
typedef __attribute__((ext_vector_type(8)))  float  v8f;

__device__ __forceinline__ v8f zero8() {
  v8f z;
#pragma unroll
  for (int i = 0; i < 8; ++i) z[i] = 0.f;
  return z;
}

__device__ __forceinline__ v8f wmma_bf16(v16bf a, v16bf b, v8f c) {
  // D = A(16x32 bf16) * B(32x16 bf16) + C(16x16 f32)
  return __builtin_amdgcn_wmma_f32_16x16x32_bf16(false, a, false, b, (short)0, c,
                                                 false, false);
}

__device__ __forceinline__ unsigned pack2bf(float a, float b) {
  const unsigned short ux = __builtin_bit_cast(unsigned short, (__bf16)a);
  const unsigned short uy = __builtin_bit_cast(unsigned short, (__bf16)b);
  return ((unsigned)uy << 16) | (unsigned)ux;
}

// ---- operand loaders -------------------------------------------------------
// A (16x32, bf16): lane&15 = row M; lanes<16 hold K {0..7,16..23}, lanes>=16
// hold K {8..15,24..31}. Per lane that is two contiguous 8-element (16 B) runs.

__device__ __forceinline__ v16bf load_A_bf16(const __bf16* src, int stride, int k0) {
  const int lane = threadIdx.x & 31;
  const int mrow = lane & 15;
  const int koff = k0 + ((lane & 16) ? 8 : 0);
  const __bf16* p = src + (size_t)mrow * stride + koff;
  union { uint4 u[2]; v16bf v; } t;
  t.u[0] = *(const uint4*)(p);        // k koff+0..7
  t.u[1] = *(const uint4*)(p + 16);   // k koff+16..23
  return t.v;
}

__device__ __forceinline__ v16bf load_A_f32(const float* src, int stride, int k0) {
  const int lane = threadIdx.x & 31;
  const int mrow = lane & 15;
  const int koff = k0 + ((lane & 16) ? 8 : 0);
  const float* p = src + (size_t)mrow * stride + koff;
  const float4 x0 = *(const float4*)(p);
  const float4 x1 = *(const float4*)(p + 4);
  const float4 x2 = *(const float4*)(p + 16);
  const float4 x3 = *(const float4*)(p + 20);
  union { unsigned u[8]; v16bf v; } t;
  t.u[0] = pack2bf(x0.x, x0.y);
  t.u[1] = pack2bf(x0.z, x0.w);
  t.u[2] = pack2bf(x1.x, x1.y);
  t.u[3] = pack2bf(x1.z, x1.w);
  t.u[4] = pack2bf(x2.x, x2.y);
  t.u[5] = pack2bf(x2.z, x2.w);
  t.u[6] = pack2bf(x3.x, x3.y);
  t.u[7] = pack2bf(x3.z, x3.w);
  return t.v;
}

// B (32x16, bf16) from f32 LDS (scan only): lane&15 = column N, K striping as A.
__device__ __forceinline__ v16bf load_B_f32(const float* src, int stride, int k0, int n0) {
  const int lane = threadIdx.x & 31;
  const int n = n0 + (lane & 15);
  const int koff = k0 + ((lane & 16) ? 8 : 0);
  const float* p = src + (size_t)koff * stride + n;
  v16bf b;
#pragma unroll
  for (int r = 0; r < 4; ++r) {
    b[2 * r]     = (__bf16)p[(size_t)(2 * r) * stride];
    b[2 * r + 1] = (__bf16)p[(size_t)(2 * r + 1) * stride];
    b[8 + 2 * r]     = (__bf16)p[(size_t)(16 + 2 * r) * stride];
    b[8 + 2 * r + 1] = (__bf16)p[(size_t)(16 + 2 * r + 1) * stride];
  }
  return b;
}

// B from pre-swizzled bf16 weights: blocks of [ktile][ntile][lane][16 elems],
// one lane = 32 contiguous bytes -> two coalesced b128 global loads.
__device__ __forceinline__ v16bf load_B_swz(const __bf16* W, int k0, int n0) {
  const int lane = threadIdx.x & 31;
  const __bf16* p =
      W + ((((size_t)(k0 >> 5) * (DD / 16)) + (n0 >> 4)) * 32 + lane) * 16;
  union { uint4 u[2]; v16bf v; } t;
  t.u[0] = *(const uint4*)(p);
  t.u[1] = *(const uint4*)(p + 8);
  return t.v;
}

// ---- kernel 0a: f32 -> bf16 (row-major, for lr_w / A-pattern use) ----------
__global__ void k_cvt_bf16(const float* __restrict__ src, __bf16* __restrict__ dst, int n) {
  int i = blockIdx.x * blockDim.x + threadIdx.x;
  if (i < n) dst[i] = (__bf16)src[i];
}

// ---- kernel 0b: f32 -> bf16 weight conversion + B-operand swizzle ----------
__global__ void k_swz_b(const float* __restrict__ src, __bf16* __restrict__ dst) {
  int idx = blockIdx.x * blockDim.x + threadIdx.x;   // over DD*DD
  if (idx >= DD * DD) return;
  const int k = idx >> 10, n = idx & (DD - 1);
  const int kt = k >> 5, klocal = k & 31;
  const int nt = n >> 4, j = n & 15;
  const int half = (klocal >> 3) & 1;                // 8..15 / 24..31 -> hi lanes
  const int lane = j + 16 * half;
  const int e = (klocal & 7) + 8 * (klocal >> 4);    // 16..31 -> elems 8..15
  dst[(((size_t)kt * (DD / 16) + nt) * 32 + lane) * 16 + e] = (__bf16)src[idx];
}

// ---- kernel 1: QKV projection + l2norm + RoPE + eta ------------------------
// grid = B*NM blocks (one 16-token mini-batch each), 256 threads (8 waves).
__global__ __launch_bounds__(256)
void k_qkv(const float* __restrict__ x,
           const __bf16* __restrict__ WqB, const __bf16* __restrict__ WkB,
           const __bf16* __restrict__ WvB, const __bf16* __restrict__ lrwB,
           const float* __restrict__ lr_b,
           float* __restrict__ XQ, float* __restrict__ XK, float* __restrict__ XV,
           float* __restrict__ eta) {
  const int blk = blockIdx.x;
  const int b = blk / NM, m = blk % NM;
  const int AS = DD + 8;
  __shared__ alignas(16) __bf16 Al[16 * (DD + 8)];
  {
    const float* src = x + ((size_t)b * LL + (size_t)m * KK) * DD;
    for (int i = threadIdx.x; i < KK * DD / 2; i += 256) {
      int t = i >> 9, d2 = i & 511;                  // pairs of columns
      const float2 f = *(const float2*)(src + (size_t)t * DD + 2 * d2);
      *(unsigned*)(&Al[t * AS + 2 * d2]) = pack2bf(f.x, f.y);
    }
  }
  __syncthreads();
  const int wave = threadIdx.x >> 5;
  const int lane = threadIdx.x & 31;
  const int nn = lane & 15;
  const int rbase = (lane & 16) ? 8 : 0;

  for (int job = wave; job < 49; job += 8) {
    if (job < 48) {
      const int mat = job >> 4;        // 0=Q,1=K,2=V
      const int h = job & 15;
      const __bf16* Wb = (mat == 0) ? WqB : (mat == 1) ? WkB : WvB;
      const int n0 = h * HDD;
      v8f acc[4];
#pragma unroll
      for (int c = 0; c < 4; ++c) acc[c] = zero8();
      // software-pipelined k-loop, 2x unrolled with even/odd operand buffers
      // (no cross-iteration register copies).
      v16bf aE = load_A_bf16(Al, AS, 0), aO;
      v16bf bE[4], bO[4];
#pragma unroll
      for (int c = 0; c < 4; ++c) bE[c] = load_B_swz(Wb, 0, n0 + c * 16);
      for (int k0 = 0; k0 < DD; k0 += 64) {
        __builtin_prefetch(
            Wb + (((size_t)((k0 + 32) >> 5) * (DD / 16)) + (n0 >> 4)) * 512, 0, 1);
        aO = load_A_bf16(Al, AS, k0 + 32);
#pragma unroll
        for (int c = 0; c < 4; ++c) bO[c] = load_B_swz(Wb, k0 + 32, n0 + c * 16);
#pragma unroll
        for (int c = 0; c < 4; ++c) acc[c] = wmma_bf16(aE, bE[c], acc[c]);
        if (k0 + 64 < DD) {
          aE = load_A_bf16(Al, AS, k0 + 64);
#pragma unroll
          for (int c = 0; c < 4; ++c) bE[c] = load_B_swz(Wb, k0 + 64, n0 + c * 16);
        }
#pragma unroll
        for (int c = 0; c < 4; ++c) acc[c] = wmma_bf16(aO, bO[c], acc[c]);
      }

      float* dst = ((mat == 0) ? XQ : (mat == 1) ? XK : XV) +
                   (((size_t)b * NHH + h) * NM + m) * (size_t)(KK * HDD);
      if (mat == 2) {
#pragma unroll
        for (int c = 0; c < 4; ++c)
#pragma unroll
          for (int v = 0; v < 8; ++v)
            dst[(size_t)(rbase + v) * HDD + c * 16 + nn] = acc[c][v];
      } else {
        // per-row (token) L2 norm over HD=64
        float scale[8];
#pragma unroll
        for (int v = 0; v < 8; ++v) {
          float ss = 0.f;
#pragma unroll
          for (int c = 0; c < 4; ++c) ss += acc[c][v] * acc[c][v];
          ss += __shfl_xor(ss, 1); ss += __shfl_xor(ss, 2);
          ss += __shfl_xor(ss, 4); ss += __shfl_xor(ss, 8);
          scale[v] = 1.f / fmaxf(sqrtf(ss), 1e-12f);
        }
        // RoPE within the 16-token mini-batch; partner d±32 lives in acc[c±2]
#pragma unroll
        for (int c = 0; c < 4; ++c) {
          const int d = c * 16 + nn;
          const float invf = powf(10000.f, -(float)(d & 31) / 32.f);
          float o[8];
#pragma unroll
          for (int v = 0; v < 8; ++v) {
            const float xc = acc[c][v] * scale[v];
            const float xr = ((c < 2) ? -acc[c + 2][v] : acc[c - 2][v]) * scale[v];
            const float ang = (float)(rbase + v) * invf;
            o[v] = xc * cosf(ang) + xr * sinf(ang);
          }
#pragma unroll
          for (int v = 0; v < 8; ++v)
            dst[(size_t)(rbase + v) * HDD + d] = o[v];
        }
      }
    } else {
      // eta logits tile: B = lr_w^T (column h = row h of lr_w -> A-loader)
      v8f acc = zero8();
      for (int k0 = 0; k0 < DD; k0 += 32) {
        v16bf a = load_A_bf16(Al, AS, k0);
        v16bf bw = load_A_bf16(lrwB, DD, k0);
        acc = wmma_bf16(a, bw, acc);
      }
      const float bias = lr_b[nn];
#pragma unroll
      for (int v = 0; v < 8; ++v) {
        const float lg = acc[v] + bias;
        const float e = (1.f / (float)HDD) / (1.f + expf(-lg));
        eta[(((size_t)b * NHH + nn) * NM + m) * KK + (rbase + v)] = e;
      }
    }
  }
}

// ---- kernel 2: sequential TTT scan -----------------------------------------
// grid = B*NH blocks, 128 threads (4 waves); wave w owns W1 columns [16w,16w+16).
__global__ __launch_bounds__(128)
void k_scan(const float* __restrict__ XQ, const float* __restrict__ XK,
            const float* __restrict__ XV, const float* __restrict__ eta,
            const float* __restrict__ W1, const float* __restrict__ b1,
            const float* __restrict__ ln_w, const float* __restrict__ ln_b,
            __bf16* __restrict__ outb) {
  const int bh = blockIdx.x;
  const int b = bh >> 4, h = bh & 15;
  __shared__ alignas(16) float W1s[64 * 68];
  __shared__ alignas(16) float xqs[16 * 68];
  __shared__ alignas(16) float xks[16 * 68];
  __shared__ alignas(16) float xvs[16 * 68];
  __shared__ alignas(16) float Zs[16 * 68];
  __shared__ alignas(16) float gr[32 * 68];      // rows 16..31 stay zero (K pad)
  __shared__ alignas(16) float Qm[16 * 36];      // cols 16..31 stay zero (K pad)
  __shared__ alignas(16) float exk[64 * 36];     // cols 16..31 stay zero (K pad)
  __shared__ float b1s[64], etas[16], lnw[64], lnb[64];

  const int tid = threadIdx.x;
  const int wave = tid >> 5, lane = tid & 31;
  const int nn = lane & 15;
  const int rbase = (lane & 16) ? 8 : 0;
  const int n0 = wave * 16;

  for (int i = tid; i < 64 * 64; i += 128) {
    int d = i >> 6, e = i & 63;
    W1s[d * 68 + e] = W1[((size_t)h * 64 + d) * 64 + e];
  }
  for (int i = tid; i < 32 * 68; i += 128) gr[i] = 0.f;
  for (int i = tid; i < 16 * 36; i += 128) Qm[i] = 0.f;
  for (int i = tid; i < 64 * 36; i += 128) exk[i] = 0.f;
  if (tid < 64) {
    b1s[tid] = b1[h * 64 + tid];
    lnw[tid] = ln_w[h * 64 + tid];
    lnb[tid] = ln_b[h * 64 + tid];
  }
  __syncthreads();

  const size_t tbase0 = (((size_t)b * NHH + h) * NM) * (size_t)(KK * HDD);

  for (int m = 0; m < NM; ++m) {
    // phase 1: stage tiles
    const size_t tb = tbase0 + (size_t)m * (KK * HDD);
    for (int i = tid; i < 16 * 64; i += 128) {
      int t = i >> 6, d = i & 63;
      xqs[t * 68 + d] = XQ[tb + i];
      xks[t * 68 + d] = XK[tb + i];
      xvs[t * 68 + d] = XV[tb + i];
    }
    if (tid < 16) etas[tid] = eta[(((size_t)b * NHH + h) * NM + m) * KK + tid];
    __syncthreads();

    // phase 2: Z1 = xk@W1 + b1 (per-wave slice); wave0: Attn = xq@xk^T
    const float b1v = b1s[n0 + nn];
    v8f accA = zero8();
    {
      v16bf a0 = load_A_f32(xks, 68, 0);
      v16bf a1 = load_A_f32(xks, 68, 32);
      v8f accz = zero8();
      accz = wmma_bf16(a0, load_B_f32(W1s, 68, 0, n0), accz);
      accz = wmma_bf16(a1, load_B_f32(W1s, 68, 32, n0), accz);
      if (wave == 0) {                   // wave-uniform branch, EXEC all-ones
        v16bf q0 = load_A_f32(xqs, 68, 0);
        v16bf q1 = load_A_f32(xqs, 68, 32);
        accA = wmma_bf16(q0, a0, accA);  // B = xk^T tile == A-load of xk
        accA = wmma_bf16(q1, a1, accA);
      }
#pragma unroll
      for (int v = 0; v < 8; ++v)
        Zs[(rbase + v) * 68 + n0 + nn] = accz[v] + b1v;
    }
    __syncthreads();

    // phase 3: grad = ln_fused_l2_bwd(Z1, xv-xk); Qm; exk
    {
      const int r = tid >> 3;
      const int j0 = (tid & 7) * 8;
      float zz[8];
#pragma unroll
      for (int j = 0; j < 8; ++j) zz[j] = Zs[r * 68 + j0 + j];
      float s = 0.f, s2 = 0.f;
#pragma unroll
      for (int j = 0; j < 8; ++j) { s += zz[j]; s2 += zz[j] * zz[j]; }
      s += __shfl_xor(s, 1); s += __shfl_xor(s, 2); s += __shfl_xor(s, 4);
      s2 += __shfl_xor(s2, 1); s2 += __shfl_xor(s2, 2); s2 += __shfl_xor(s2, 4);
      const float mu = s * (1.f / 64.f);
      const float var = s2 * (1.f / 64.f) - mu * mu;
      const float isd = rsqrtf(var + EPSF);
      float gy[8], xh[8], sg = 0.f, sgx = 0.f;
#pragma unroll
      for (int j = 0; j < 8; ++j) {
        const int d = j0 + j;
        xh[j] = (zz[j] - mu) * isd;
        const float y = lnw[d] * xh[j] + lnb[d];
        const float tg = xvs[r * 68 + d] - xks[r * 68 + d];
        gy[j] = (y - tg) * lnw[d];
        sg += gy[j]; sgx += gy[j] * xh[j];
      }
      sg += __shfl_xor(sg, 1); sg += __shfl_xor(sg, 2); sg += __shfl_xor(sg, 4);
      sgx += __shfl_xor(sgx, 1); sgx += __shfl_xor(sgx, 2); sgx += __shfl_xor(sgx, 4);
#pragma unroll
      for (int j = 0; j < 8; ++j)
        gr[r * 68 + j0 + j] = (64.f * gy[j] - sg - xh[j] * sgx) * isd * (1.f / 64.f);
    }
    if (wave == 0) {  // Qm[i][j] = (j<=i) ? eta[j]*(Attn[i][j]+1) : 0
#pragma unroll
      for (int v = 0; v < 8; ++v) {
        const int i = rbase + v;
        Qm[i * 36 + nn] = (nn <= i) ? etas[nn] * (accA[v] + 1.f) : 0.f;
      }
    }
    for (int i = tid; i < 64 * 16; i += 128) {
      int d = i >> 4, t = i & 15;
      exk[d * 36 + t] = etas[t] * xks[t * 68 + d];
    }
    __syncthreads();

    // phase 4: Z1_bar = xq@W1 + b1 - Qm@grad
    {
      v16bf q0 = load_A_f32(xqs, 68, 0);
      v16bf q1 = load_A_f32(xqs, 68, 32);
      v8f accb = zero8();
      accb = wmma_bf16(q0, load_B_f32(W1s, 68, 0, n0), accb);
      accb = wmma_bf16(q1, load_B_f32(W1s, 68, 32, n0), accb);
      v8f accs = zero8();
      accs = wmma_bf16(load_A_f32(Qm, 36, 0), load_B_f32(gr, 68, 0, n0), accs);
#pragma unroll
      for (int v = 0; v < 8; ++v)
        Zs[(rbase + v) * 68 + n0 + nn] = accb[v] + b1v - accs[v];
    }
    __syncthreads();

    // phase 5: state update + output
    {
      v16bf bg = load_B_f32(gr, 68, 0, n0);
#pragma unroll
      for (int mt = 0; mt < 4; ++mt) {
        v8f u = zero8();
        u = wmma_bf16(load_A_f32(exk + mt * 16 * 36, 36, 0), bg, u);
#pragma unroll
        for (int v = 0; v < 8; ++v)
          W1s[(mt * 16 + rbase + v) * 68 + n0 + nn] -= u[v];
      }
    }
    if (tid < 64) {
      float s = 0.f;
#pragma unroll
      for (int t = 0; t < 16; ++t) s += etas[t] * gr[t * 68 + tid];
      b1s[tid] -= s;
    }
    {
      const int r = tid >> 3;
      const int j0 = (tid & 7) * 8;
      float zz[8];
#pragma unroll
      for (int j = 0; j < 8; ++j) zz[j] = Zs[r * 68 + j0 + j];
      float s = 0.f, s2 = 0.f;
#pragma unroll
      for (int j = 0; j < 8; ++j) { s += zz[j]; s2 += zz[j] * zz[j]; }
      s += __shfl_xor(s, 1); s += __shfl_xor(s, 2); s += __shfl_xor(s, 4);
      s2 += __shfl_xor(s2, 1); s2 += __shfl_xor(s2, 2); s2 += __shfl_xor(s2, 4);
      const float mu = s * (1.f / 64.f);
      const float var = s2 * (1.f / 64.f) - mu * mu;
      const float isd = rsqrtf(var + EPSF);
      __bf16* op = outb + ((size_t)b * LL + (size_t)m * KK + r) * DD + h * HDD;
#pragma unroll
      for (int j = 0; j < 8; ++j) {
        const int d = j0 + j;
        const float y = lnw[d] * ((zz[j] - mu) * isd) + lnb[d];
        op[d] = (__bf16)(xqs[r * 68 + d] + y);
      }
    }
    __syncthreads();
  }
}

// ---- kernel 3: output projection + final LayerNorm -------------------------
// grid = B*L/16 blocks, 512 threads (16 waves); wave w owns 64 output columns.
__global__ __launch_bounds__(512)
void k_out(const __bf16* __restrict__ ob, const __bf16* __restrict__ WoB,
           const float* __restrict__ pn_w, const float* __restrict__ pn_b,
           float* __restrict__ out) {
  const int rb = blockIdx.x;
  const int AS = DD + 8;
  __shared__ alignas(16) __bf16 Al[16 * (DD + 8)];
  __shared__ float ps[16][16], ps2[16][16], mus[16], isds[16];
  {
    const unsigned* src = (const unsigned*)(ob + (size_t)rb * KK * DD);
    for (int i = threadIdx.x; i < KK * DD / 2; i += 512) {
      int t = i >> 9, d2 = i & 511;
      *(unsigned*)(&Al[t * AS + 2 * d2]) = src[i];
    }
  }
  __syncthreads();
  const int wave = threadIdx.x >> 5, lane = threadIdx.x & 31;
  const int nn = lane & 15;
  const int rbase = (lane & 16) ? 8 : 0;
  const int nb = wave * 64;
  v8f acc[4];
#pragma unroll
  for (int c = 0; c < 4; ++c) acc[c] = zero8();
  // software-pipelined k-loop, 2x unrolled with even/odd operand buffers
  v16bf aE = load_A_bf16(Al, AS, 0), aO;
  v16bf bE[4], bO[4];
#pragma unroll
  for (int c = 0; c < 4; ++c) bE[c] = load_B_swz(WoB, 0, nb + c * 16);
  for (int k0 = 0; k0 < DD; k0 += 64) {
    __builtin_prefetch(
        WoB + (((size_t)((k0 + 32) >> 5) * (DD / 16)) + (nb >> 4)) * 512, 0, 1);
    aO = load_A_bf16(Al, AS, k0 + 32);
#pragma unroll
    for (int c = 0; c < 4; ++c) bO[c] = load_B_swz(WoB, k0 + 32, nb + c * 16);
#pragma unroll
    for (int c = 0; c < 4; ++c) acc[c] = wmma_bf16(aE, bE[c], acc[c]);
    if (k0 + 64 < DD) {
      aE = load_A_bf16(Al, AS, k0 + 64);
#pragma unroll
      for (int c = 0; c < 4; ++c) bE[c] = load_B_swz(WoB, k0 + 64, nb + c * 16);
    }
#pragma unroll
    for (int c = 0; c < 4; ++c) acc[c] = wmma_bf16(aO, bO[c], acc[c]);
  }

  // per-wave partial row sums -> cross-wave LN reduction in LDS
  float sp[8], sq[8];
#pragma unroll
  for (int v = 0; v < 8; ++v) {
    float s = 0.f, s2 = 0.f;
#pragma unroll
    for (int c = 0; c < 4; ++c) { const float xv = acc[c][v]; s += xv; s2 += xv * xv; }
    s += __shfl_xor(s, 1); s += __shfl_xor(s, 2); s += __shfl_xor(s, 4); s += __shfl_xor(s, 8);
    s2 += __shfl_xor(s2, 1); s2 += __shfl_xor(s2, 2); s2 += __shfl_xor(s2, 4); s2 += __shfl_xor(s2, 8);
    sp[v] = s; sq[v] = s2;
  }
  if (nn == 0) {
#pragma unroll
    for (int v = 0; v < 8; ++v) { ps[wave][rbase + v] = sp[v]; ps2[wave][rbase + v] = sq[v]; }
  }
  __syncthreads();
  if (threadIdx.x < 16) {
    float s = 0.f, s2 = 0.f;
    for (int w = 0; w < 16; ++w) { s += ps[w][threadIdx.x]; s2 += ps2[w][threadIdx.x]; }
    const float mu = s * (1.f / (float)DD);
    const float var = s2 * (1.f / (float)DD) - mu * mu;
    mus[threadIdx.x] = mu;
    isds[threadIdx.x] = rsqrtf(var + EPSF);
  }
  __syncthreads();
  float* dst = out + (size_t)rb * KK * DD;
#pragma unroll
  for (int v = 0; v < 8; ++v) {
    const int row = rbase + v;
    const float mu = mus[row], isd = isds[row];
#pragma unroll
    for (int c = 0; c < 4; ++c) {
      const int d = nb + c * 16 + nn;
      dst[(size_t)row * DD + d] = (acc[c][v] - mu) * isd * pn_w[d] + pn_b[d];
    }
  }
}

// ---- host-side launch ------------------------------------------------------
extern "C" void kernel_launch(void* const* d_in, const int* in_sizes, int n_in,
                              void* d_out, int out_size, void* d_ws, size_t ws_size,
                              hipStream_t stream) {
  const float* x    = (const float*)d_in[0];
  const float* Wq   = (const float*)d_in[1];
  const float* Wk   = (const float*)d_in[2];
  const float* Wv   = (const float*)d_in[3];
  const float* Wo   = (const float*)d_in[4];
  const float* W1   = (const float*)d_in[5];
  const float* b1   = (const float*)d_in[6];
  const float* ln_w = (const float*)d_in[7];
  const float* ln_b = (const float*)d_in[8];
  const float* lr_w = (const float*)d_in[9];
  const float* lr_b = (const float*)d_in[10];
  const float* pn_w = (const float*)d_in[11];
  const float* pn_b = (const float*)d_in[12];

  const size_t NQ = (size_t)BB * NHH * NM * KK * HDD;   // 16,777,216 floats
  const size_t NE = (size_t)BB * NHH * NM * KK;         // 262,144 floats
  const size_t NO = (size_t)BB * LL * DD;               // 16,777,216 elems

  float* XQ  = (float*)d_ws;
  float* XK  = XQ + NQ;
  float* XV  = XK + NQ;
  float* eta = XV + NQ;
  __bf16* outb = (__bf16*)(eta + NE);
  __bf16* WqB  = outb + NO;                 // swizzled B-operand layout
  __bf16* WkB  = WqB + (size_t)DD * DD;
  __bf16* WvB  = WkB + (size_t)DD * DD;
  __bf16* WoB  = WvB + (size_t)DD * DD;
  __bf16* lrwB = WoB + (size_t)DD * DD;     // row-major bf16 (A-pattern)

  const int ncv = DD * DD;
  k_swz_b<<<(ncv + 255) / 256, 256, 0, stream>>>(Wq, WqB);
  k_swz_b<<<(ncv + 255) / 256, 256, 0, stream>>>(Wk, WkB);
  k_swz_b<<<(ncv + 255) / 256, 256, 0, stream>>>(Wv, WvB);
  k_swz_b<<<(ncv + 255) / 256, 256, 0, stream>>>(Wo, WoB);
  k_cvt_bf16<<<(NHH * DD + 255) / 256, 256, 0, stream>>>(lr_w, lrwB, NHH * DD);

  k_qkv<<<BB * NM, 256, 0, stream>>>(x, WqB, WkB, WvB, lrwB, lr_b, XQ, XK, XV, eta);
  k_scan<<<BB * NHH, 128, 0, stream>>>(XQ, XK, XV, eta, W1, b1, ln_w, ln_b, outb);
  k_out<<<BB * LL / KK, 512, 0, stream>>>(outb, WoB, pn_w, pn_b, (float*)d_out);
}
